// GraphTransformerPool_62457414418904
// MI455X (gfx1250) — compile-verified, matrix operand order checked by reading, not run on hardware
//
#include <hip/hip_runtime.h>

// ---------------------------------------------------------------------------
// Types for CDNA5 WMMA (wave32)
// ---------------------------------------------------------------------------
typedef __bf16 v16bf __attribute__((ext_vector_type(16)));
typedef float  v8f   __attribute__((ext_vector_type(8)));
typedef unsigned int v4u __attribute__((ext_vector_type(4)));

#define NB   64
#define SEQ  513          // L+1
#define DIM  256
#define NH   8
#define DH   32
#define DFF  1024
#define NLAY 4
#define NNODES 16384
#define BS   (NB * SEQ)   // 32832 rows, divisible by 64
#define KPAD 544          // 34 * 16 key tiles (covers 513)

__device__ __forceinline__ unsigned short f2bf_bits(float f) {
    union { float f; unsigned u; } x; x.f = f;
    unsigned r = x.u + 0x7FFFu + ((x.u >> 16) & 1u);   // round-to-nearest-even
    return (unsigned short)(r >> 16);
}

// Load a 16-row x 32-K bf16 fragment strip: lane holds K = {khalf..+7, 16+khalf..+7}
// p must point at (row_base + khalf), 16B aligned.
__device__ __forceinline__ v16bf frag_load(const __bf16* p) {
    union { v16bf v; v4u u[2]; } f;
    f.u[0] = *(const v4u*)(p);
    f.u[1] = *(const v4u*)(p + 16);
    return f.v;
}

__device__ __forceinline__ v8f wmma_bf16(v16bf a, v16bf b, v8f c) {
    return __builtin_amdgcn_wmma_f32_16x16x32_bf16(false, a, false, b, (short)0, c, false, false);
}

// ---------------------------------------------------------------------------
// Utility kernels
// ---------------------------------------------------------------------------
__global__ __launch_bounds__(256) void cvt_f32_bf16(const float* __restrict__ s,
                                                    unsigned short* __restrict__ d, int n) {
    int i = blockIdx.x * 256 + threadIdx.x;
    if (i < n) d[i] = f2bf_bits(s[i]);
}

__global__ __launch_bounds__(64) void first_kernel(const int* __restrict__ batch,
                                                   int* __restrict__ first, int n) {
    int b = threadIdx.x;
    int lo = 0, hi = n;
    while (lo < hi) { int mid = (lo + hi) >> 1; if (batch[mid] < b) lo = mid + 1; else hi = mid; }
    first[b] = lo;
}

__global__ __launch_bounds__(256) void init_x_kernel(float* __restrict__ xf,
                                                     int* __restrict__ kv,
                                                     const float* __restrict__ cls) {
    int row = blockIdx.x;             // 0..BS-1
    int c   = threadIdx.x;
    int s   = row % SEQ;
    xf[(size_t)row * DIM + c] = (s == 0) ? cls[c] : 0.0f;
    if (c == 0) kv[row] = (s == 0) ? 1 : 0;
}

__global__ __launch_bounds__(256) void scatter_kernel(const float* __restrict__ h,
                                                      const int* __restrict__ batch,
                                                      const int* __restrict__ first,
                                                      float* __restrict__ xf,
                                                      int* __restrict__ kv) {
    int i = blockIdx.x;               // node
    int c = threadIdx.x;
    int b = batch[i];
    int s = 1 + (i - first[b]);
    if (s < SEQ) {
        xf[((size_t)b * SEQ + s) * DIM + c] = h[(size_t)i * DIM + c];
        if (c == 0) kv[b * SEQ + s] = 1;
    }
}

__global__ __launch_bounds__(256) void final_kernel(const float* __restrict__ xf,
                                                    float* __restrict__ out) {
    int b = blockIdx.x, c = threadIdx.x;
    out[(size_t)b * DIM + c] = xf[(size_t)b * SEQ * DIM + c];
}

// ---------------------------------------------------------------------------
// Transpose V out of qkv into vT[b, h, dh, key] (bf16, keys zero-padded to KPAD)
// block = (b*NH + h); 256 threads = 32 dh x 8 key-phases.
// ---------------------------------------------------------------------------
__global__ __launch_bounds__(256) void vtrans_kernel(const unsigned short* __restrict__ qkv,
                                                     unsigned short* __restrict__ vT) {
    int bh = blockIdx.x;                 // 0 .. NB*NH-1
    int b = bh / NH, h = bh % NH;
    int dh = threadIdx.x & 31;
    int k0 = threadIdx.x >> 5;           // 0..7
    const unsigned short* src = qkv + (size_t)b * SEQ * 768 + 512 + h * 32 + dh;
    unsigned short* dst = vT + ((size_t)bh * DH + dh) * KPAD;
    for (int key = k0; key < KPAD; key += 8)
        dst[key] = (key < SEQ) ? src[(size_t)key * 768] : (unsigned short)0;
}

// ---------------------------------------------------------------------------
// Generic bf16 WMMA GEMM:  C[M,N] = act(A[M,K] . W[N,K]^T + bias[N])
// Block: 256 thr = 8 waves; wave tile 32x64 (2x4 WMMA); block tile 64M x 256N.
// M % 64 == 0, N % 256 == 0, K % 32 == 0 (true for every GEMM here).
// ---------------------------------------------------------------------------
__global__ __launch_bounds__(256) void gemm_bf16(const __bf16* __restrict__ A,
                                                 const __bf16* __restrict__ W,
                                                 const float* __restrict__ bias,
                                                 unsigned short* __restrict__ outb,
                                                 float* __restrict__ outf,
                                                 int N, int K, int relu) {
    int lane = threadIdx.x & 31;
    int wid  = threadIdx.x >> 5;
    int wm   = wid & 1;                  // 2 waves in M
    int wn   = wid >> 1;                 // 4 waves in N
    int m0 = blockIdx.x * 64  + wm * 32;
    int n0 = blockIdx.y * 256 + wn * 64;
    int khalf = (lane >> 4) * 8;
    int l15   = lane & 15;

    const __bf16* pa0 = A + (size_t)(m0 + l15)      * K + khalf;
    const __bf16* pa1 = A + (size_t)(m0 + 16 + l15) * K + khalf;
    const __bf16* pb0 = W + (size_t)(n0 + l15)      * K + khalf;
    const __bf16* pb1 = W + (size_t)(n0 + 16 + l15) * K + khalf;
    const __bf16* pb2 = W + (size_t)(n0 + 32 + l15) * K + khalf;
    const __bf16* pb3 = W + (size_t)(n0 + 48 + l15) * K + khalf;

    v8f acc[2][4] = {};
    for (int k0 = 0; k0 < K; k0 += 32) {
        v16bf a0 = frag_load(pa0 + k0);
        v16bf a1 = frag_load(pa1 + k0);
        v16bf b0 = frag_load(pb0 + k0);
        v16bf b1 = frag_load(pb1 + k0);
        v16bf b2 = frag_load(pb2 + k0);
        v16bf b3 = frag_load(pb3 + k0);
        acc[0][0] = wmma_bf16(a0, b0, acc[0][0]);
        acc[0][1] = wmma_bf16(a0, b1, acc[0][1]);
        acc[0][2] = wmma_bf16(a0, b2, acc[0][2]);
        acc[0][3] = wmma_bf16(a0, b3, acc[0][3]);
        acc[1][0] = wmma_bf16(a1, b0, acc[1][0]);
        acc[1][1] = wmma_bf16(a1, b1, acc[1][1]);
        acc[1][2] = wmma_bf16(a1, b2, acc[1][2]);
        acc[1][3] = wmma_bf16(a1, b3, acc[1][3]);
    }

#pragma unroll
    for (int tm = 0; tm < 2; ++tm) {
#pragma unroll
        for (int tn = 0; tn < 4; ++tn) {
            int col = n0 + tn * 16 + l15;
            float bv = bias[col];
#pragma unroll
            for (int r = 0; r < 8; ++r) {
                int row = m0 + tm * 16 + r + 8 * (lane >> 4);
                float v = acc[tm][tn][r] + bv;
                if (relu) v = fmaxf(v, 0.0f);
                size_t idx = (size_t)row * N + col;
                if (outf) outf[idx] = v;
                if (outb) outb[idx] = f2bf_bits(v);
            }
        }
    }
}

// ---------------------------------------------------------------------------
// Attention: one wave per (query-tile, head, batch).
// qkv layout: [BS, 768] bf16 (q | k | v blocks of 256, head h at h*32).
// vT layout:  [NB*NH, DH, KPAD] bf16 (zero-padded keys).
// ---------------------------------------------------------------------------
__global__ __launch_bounds__(32) void attn_kernel(const __bf16* __restrict__ qkv,
                                                  const __bf16* __restrict__ vT,
                                                  const int* __restrict__ kv,
                                                  unsigned short* __restrict__ ob) {
    __shared__ float          sc[16][KPAD];
    __shared__ unsigned short at[16][KPAD];

    const int qt = blockIdx.x, h = blockIdx.y, b = blockIdx.z;
    const int lane  = threadIdx.x;
    const int l15   = lane & 15;
    const int khalf = (lane >> 4) * 8;
    const int q0 = qt * 16;
    const float scale = 0.17677669529663687f;  // 1/sqrt(32)
    const size_t rowbase = (size_t)b * SEQ * 768;

    // A fragment: q rows (clamped; padded rows discarded at the end)
    int qr = q0 + l15; if (qr > SEQ - 1) qr = SEQ - 1;
    v16bf qa = frag_load(qkv + rowbase + (size_t)qr * 768 + h * 32 + khalf);

    // scores = q . k^T, scaled + masked -> LDS
    for (int kt = 0; kt < KPAD / 16; ++kt) {
        int col = kt * 16 + l15;
        int kc  = col; if (kc > SEQ - 1) kc = SEQ - 1;
        v16bf kb = frag_load(qkv + rowbase + (size_t)kc * 768 + 256 + h * 32 + khalf);
        v8f d = {};
        d = wmma_bf16(qa, kb, d);
        bool valid = (col < SEQ) && (kv[b * SEQ + col] != 0);
#pragma unroll
        for (int r = 0; r < 8; ++r) {
            int row = r + 8 * (lane >> 4);
            sc[row][col] = valid ? d[r] * scale : -1.0e9f;
        }
    }
    __syncthreads();

    // row softmax (lanes 0..15 each own a query row), requantize to bf16
    if (lane < 16) {
        float m = -3.0e38f;
        for (int j = 0; j < KPAD; ++j) m = fmaxf(m, sc[lane][j]);
        float s = 0.0f;
        for (int j = 0; j < KPAD; ++j) { float e = __expf(sc[lane][j] - m); sc[lane][j] = e; s += e; }
        float inv = 1.0f / s;
        for (int j = 0; j < KPAD; ++j) at[lane][j] = f2bf_bits(sc[lane][j] * inv);
    }
    __syncthreads();

    // o = attn . v  via transposed V (padded keys have weight exactly 0)
    v8f acc0 = {}, acc1 = {};
    const __bf16* vt0 = vT + ((size_t)(b * NH + h) * DH + l15)      * KPAD + khalf;
    const __bf16* vt1 = vT + ((size_t)(b * NH + h) * DH + 16 + l15) * KPAD + khalf;
    for (int kt = 0; kt < KPAD / 32; ++kt) {
        v16bf af = frag_load((const __bf16*)&at[l15][kt * 32 + khalf]);
        v16bf v0 = frag_load(vt0 + kt * 32);
        v16bf v1 = frag_load(vt1 + kt * 32);
        acc0 = wmma_bf16(af, v0, acc0);
        acc1 = wmma_bf16(af, v1, acc1);
    }

    // store o[16, 32] tile -> ob[BS, 256] at columns h*32 + dh
#pragma unroll
    for (int r = 0; r < 8; ++r) {
        int row = r + 8 * (lane >> 4);
        int s = q0 + row;
        if (s < SEQ) {
            size_t base = ((size_t)b * SEQ + s) * DIM + h * 32;
            ob[base + l15]      = f2bf_bits(acc0[r]);
            ob[base + 16 + l15] = f2bf_bits(acc1[r]);
        }
    }
}

// ---------------------------------------------------------------------------
// Residual + LayerNorm over DIM=256; writes f32 and bf16 copies of x.
// ---------------------------------------------------------------------------
__global__ __launch_bounds__(256) void ln_kernel(float* __restrict__ x,
                                                 const float* __restrict__ t,
                                                 const float* __restrict__ g,
                                                 const float* __restrict__ bt,
                                                 unsigned short* __restrict__ xb) {
    __shared__ float red[256];
    int row = blockIdx.x, c = threadIdx.x;
    size_t idx = (size_t)row * DIM + c;
    float v = x[idx] + t[idx];

    red[c] = v; __syncthreads();
    for (int s = 128; s > 0; s >>= 1) { if (c < s) red[c] += red[c + s]; __syncthreads(); }
    float mu = red[0] * (1.0f / DIM);
    __syncthreads();

    float d = v - mu;
    red[c] = d * d; __syncthreads();
    for (int s = 128; s > 0; s >>= 1) { if (c < s) red[c] += red[c + s]; __syncthreads(); }
    float var = red[0] * (1.0f / DIM);

    float y = d * rsqrtf(var + 1e-5f) * g[c] + bt[c];
    x[idx]  = y;
    xb[idx] = f2bf_bits(y);
}

// ---------------------------------------------------------------------------
// Host: workspace layout + launch sequence
// ---------------------------------------------------------------------------
extern "C" void kernel_launch(void* const* d_in, const int* in_sizes, int n_in,
                              void* d_out, int out_size, void* d_ws, size_t ws_size,
                              hipStream_t stream) {
    (void)in_sizes; (void)n_in; (void)out_size; (void)ws_size;
    const float* h_nodes = (const float*)d_in[0];
    const int*   batch   = (const int*)  d_in[1];
    const float* cls     = (const float*)d_in[2];
    const float* Wqkv    = (const float*)d_in[3];
    const float* bqkv    = (const float*)d_in[4];
    const float* Wo      = (const float*)d_in[5];
    const float* bo      = (const float*)d_in[6];
    const float* W1      = (const float*)d_in[7];
    const float* b1      = (const float*)d_in[8];
    const float* W2      = (const float*)d_in[9];
    const float* b2      = (const float*)d_in[10];
    const float* ln1_g   = (const float*)d_in[11];
    const float* ln1_b   = (const float*)d_in[12];
    const float* ln2_g   = (const float*)d_in[13];
    const float* ln2_b   = (const float*)d_in[14];

    char* ws = (char*)d_ws;
    size_t off = 0;
    auto alloc = [&](size_t bytes) { char* p = ws + off; off += (bytes + 255) & ~size_t(255); return p; };

    float*          xf   = (float*)         alloc((size_t)BS * DIM * 4);
    unsigned short* xb   = (unsigned short*)alloc((size_t)BS * DIM * 2);
    unsigned short* qkvb = (unsigned short*)alloc((size_t)BS * 768 * 2);
    unsigned short* vT   = (unsigned short*)alloc((size_t)NB * NH * DH * KPAD * 2);
    unsigned short* obuf = (unsigned short*)alloc((size_t)BS * DIM * 2);
    unsigned short* ffh  = (unsigned short*)alloc((size_t)BS * DFF * 2);
    float*          tf   = (float*)         alloc((size_t)BS * DIM * 4);
    unsigned short* WqkvB= (unsigned short*)alloc((size_t)NLAY * 3 * DIM * DIM * 2);
    unsigned short* WoB  = (unsigned short*)alloc((size_t)NLAY * DIM * DIM * 2);
    unsigned short* W1B  = (unsigned short*)alloc((size_t)NLAY * DFF * DIM * 2);
    unsigned short* W2B  = (unsigned short*)alloc((size_t)NLAY * DIM * DFF * 2);
    int*            kv   = (int*)           alloc((size_t)BS * 4);
    int*            first= (int*)           alloc(64 * 4);

    // 1) weights -> bf16
    int nWqkv = NLAY * 3 * DIM * DIM, nWo = NLAY * DIM * DIM;
    int nW1 = NLAY * DFF * DIM, nW2 = NLAY * DIM * DFF;
    cvt_f32_bf16<<<(nWqkv + 255) / 256, 256, 0, stream>>>(Wqkv, WqkvB, nWqkv);
    cvt_f32_bf16<<<(nWo   + 255) / 256, 256, 0, stream>>>(Wo,   WoB,   nWo);
    cvt_f32_bf16<<<(nW1   + 255) / 256, 256, 0, stream>>>(W1,   W1B,   nW1);
    cvt_f32_bf16<<<(nW2   + 255) / 256, 256, 0, stream>>>(W2,   W2B,   nW2);

    // 2) dense batch build
    first_kernel<<<1, 64, 0, stream>>>(batch, first, NNODES);
    init_x_kernel<<<BS, 256, 0, stream>>>(xf, kv, cls);
    scatter_kernel<<<NNODES, 256, 0, stream>>>(h_nodes, batch, first, xf, kv);
    cvt_f32_bf16<<<((size_t)BS * DIM + 255) / 256, 256, 0, stream>>>(xf, xb, BS * DIM);

    dim3 gThr(256);
    for (int l = 0; l < NLAY; ++l) {
        const __bf16* wq = (const __bf16*)(WqkvB + (size_t)l * 3 * DIM * DIM);
        const __bf16* wo = (const __bf16*)(WoB   + (size_t)l * DIM * DIM);
        const __bf16* w1 = (const __bf16*)(W1B   + (size_t)l * DFF * DIM);
        const __bf16* w2 = (const __bf16*)(W2B   + (size_t)l * DIM * DFF);

        // QKV: [BS,256] x [768,256]^T -> bf16
        gemm_bf16<<<dim3(BS / 64, 768 / 256), gThr, 0, stream>>>(
            (const __bf16*)xb, wq, bqkv + (size_t)l * 768, qkvb, nullptr, 768, DIM, 0);

        // V transpose for the attn.V matmul
        vtrans_kernel<<<NB * NH, 256, 0, stream>>>(qkvb, vT);

        // attention
        attn_kernel<<<dim3((SEQ + 15) / 16, NH, NB), dim3(32), 0, stream>>>(
            (const __bf16*)qkvb, (const __bf16*)vT, kv, obuf);

        // O-proj -> f32, then residual+LN1
        gemm_bf16<<<dim3(BS / 64, 256 / 256), gThr, 0, stream>>>(
            (const __bf16*)obuf, wo, bo + (size_t)l * DIM, nullptr, tf, DIM, DIM, 0);
        ln_kernel<<<BS, 256, 0, stream>>>(xf, tf, ln1_g + (size_t)l * DIM,
                                          ln1_b + (size_t)l * DIM, xb);

        // FFN1 (ReLU) -> bf16
        gemm_bf16<<<dim3(BS / 64, DFF / 256), gThr, 0, stream>>>(
            (const __bf16*)xb, w1, b1 + (size_t)l * DFF, ffh, nullptr, DFF, DIM, 1);

        // FFN2 -> f32, then residual+LN2
        gemm_bf16<<<dim3(BS / 64, 256 / 256), gThr, 0, stream>>>(
            (const __bf16*)ffh, w2, b2 + (size_t)l * DIM, nullptr, tf, DIM, DFF, 0);
        ln_kernel<<<BS, 256, 0, stream>>>(xf, tf, ln2_g + (size_t)l * DIM,
                                          ln2_b + (size_t)l * DIM, xb);
    }

    // CLS rows -> output [64, 256] f32
    final_kernel<<<NB, 256, 0, stream>>>(xf, (float*)d_out);
}
